// FCPairedLayer_5162550689973
// MI455X (gfx1250) — matrix operand order repaired and checked by reading, not compiled, not run on hardware
//
#include <hip/hip_runtime.h>
#include <hip/hip_bf16.h>
#include <math.h>

// FCPairedLayer, restructured:
//  Stage 1 (WMMA GEMM): six 512x64 @ 64x256 GEMMs producing row tables
//      P[0]=A0 (xl*W1[0:64])    P[1]=B0 (xr*W1[64:128])
//      P[2]=A1 (xl*W1[128:192]) P[3]=B1 (xr*W1[192:256])
//      P[4]=A2 (xl*W1[256:320]) P[5]=B2 (xr*W1[320:384])
//  Stage 2 (VALU): h[i,j] = A0[i]+B0[j]+b1 + mu*(A1[i-1]+B1[j+1]) + md*(A2[i+1]+B2[j-1])
//      then LayerNorm -> CELU -> dot(W2) + b2 per (i,j), wave32 reductions.

typedef __attribute__((ext_vector_type(16))) _Float16 v16h;
typedef __attribute__((ext_vector_type(8)))  float    v8f;

#define NN   512
#define HID  256

// ---------------- Stage 1: 6x (M=512, K=64, N=256) GEMM via v_wmma_f32_16x16x32_f16 ----
// One wave per 16x16 C tile. Tiles: kb(6) x mt(32) x nt(16) = 3072 waves = 768 blocks x 4 waves.
__global__ __launch_bounds__(128) void fcpaired_stage1_wmma(
    const float* __restrict__ xl, const float* __restrict__ xr,
    const float* __restrict__ W1, float* __restrict__ P) {
  const int lane = threadIdx.x & 31;
  const int wave = blockIdx.x * 4 + (threadIdx.x >> 5);

  const int kb  = wave >> 9;        // 0..5  (feature block)
  const int rem = wave & 511;
  const int mt  = rem >> 4;         // 0..31 (M tile)
  const int nt  = rem & 15;         // 0..15 (N tile)
  const int mbase = mt * 16;
  const int nbase = nt * 16;
  const int fofs  = kb * 64;        // row offset into W1 (W1 is [384][256] row-major)
  const float* X  = (kb & 1) ? xr : xl;   // even blocks use xl, odd use xr

  const int lo = lane & 15;
  const int hi = lane >> 4;

  v8f c = {};
  const float* xrow = X + (mbase + lo) * 64;   // A row M = mbase + (lane&15)

  #pragma unroll
  for (int K0 = 0; K0 < 64; K0 += 32) {
    v16h a, b;
    #pragma unroll
    for (int e = 0; e < 16; ++e) {
      // 16-bit A 16x32 layout (ISA 7.12.2): lanes 0-15 rows M, VGPR0..3 -> K=0..7,
      // VGPR4..7 -> K=16..23; lanes 16-31 get K+8.  e = 2*vgpr + half.
      int kA = K0 + ((e >> 3) << 4) + (((e >> 1) & 3) << 1) + (e & 1) + (hi << 3);
      a[e] = (_Float16)xrow[kA];
      // 16-bit B 32x16 layout: lane holds column N=lane&15; same K order as A:
      // e<8 -> K=8*hi+e, e>=8 -> K=16+8*hi+(e-8).
      int kB = K0 + ((e >> 3) << 4) + (hi << 3) + (e & 7);
      b[e] = (_Float16)W1[(fofs + kB) * HID + (nbase + lo)];
    }
    // 8 args: (neg_a, A, neg_b, B, c_mod, C, reuse_a, reuse_b)
    c = __builtin_amdgcn_wmma_f32_16x16x32_f16(false, a, false, b, (short)0, c,
                                               false, false);
  }

  // 32-bit C/D 16x16 layout: VGPR v -> M = v + 8*(lane>=16), N = lane&15.
  float* Pk = P + kb * (NN * HID);
  #pragma unroll
  for (int v = 0; v < 8; ++v) {
    int row = mbase + v + (hi << 3);
    Pk[row * HID + (nbase + lo)] = c[v];
  }
}

// ---------------- Stage 2: per-(i,j) combine + LayerNorm + CELU + dot(W2) ----------------
// One wave per (i, 64-wide j chunk). 512 rows * 8 chunks = 4096 waves = 512 blocks x 8 waves.
__global__ __launch_bounds__(256) void fcpaired_stage2(
    const float* __restrict__ P,  const float* __restrict__ b1,
    const float* __restrict__ gamma, const float* __restrict__ beta,
    const float* __restrict__ W2, const float* __restrict__ b2,
    float* __restrict__ out) {
  const int lane = threadIdx.x & 31;
  const int wave = blockIdx.x * 8 + (threadIdx.x >> 5);
  const int i  = wave >> 3;
  const int jb = (wave & 7) * 64;

  const int im1 = (i > 0)      ? i - 1 : 0;     // clamped; masked out when i==0
  const int ip1 = (i < NN - 1) ? i + 1 : NN - 1;

  const float* A0 = P + (0 * NN + i)   * HID;
  const float* A1 = P + (2 * NN + im1) * HID;
  const float* A2 = P + (4 * NN + ip1) * HID;

  // Per-lane hidden slice: t = lane + 32*r  (coalesced 128B/wave loads)
  float a0[8], a1[8], a2[8], bb[8], gg[8], bt[8], w2[8];
  #pragma unroll
  for (int r = 0; r < 8; ++r) {
    int t = lane + 32 * r;
    a0[r] = A0[t]; a1[r] = A1[t]; a2[r] = A2[t];
    bb[r] = b1[t]; gg[r] = gamma[t]; bt[r] = beta[t]; w2[r] = W2[t];
  }
  const float b2s = b2[0];

  for (int j = jb; j < jb + 64; ++j) {
    const float mu_m = (i >= 1 && j <= NN - 2) ? 1.0f : 0.0f;  // up-shift valid
    const float md_m = (i <= NN - 2 && j >= 1) ? 1.0f : 0.0f;  // down-shift valid
    const int ju = (j < NN - 1) ? j + 1 : NN - 1;
    const int jd = (j > 0)      ? j - 1 : 0;
    const float* B0 = P + (1 * NN + j)  * HID;
    const float* B1 = P + (3 * NN + ju) * HID;
    const float* B2 = P + (5 * NN + jd) * HID;

    float h[8];
    float s = 0.0f, s2 = 0.0f;
    #pragma unroll
    for (int r = 0; r < 8; ++r) {
      int t = lane + 32 * r;
      float v = a0[r] + B0[t] + bb[r]
              + mu_m * (a1[r] + B1[t])
              + md_m * (a2[r] + B2[t]);
      h[r] = v;
      s  += v;
      s2 += v * v;
    }
    // wave32 tree reduction for LN statistics
    #pragma unroll
    for (int m = 16; m >= 1; m >>= 1) {
      s  += __shfl_xor(s,  m, 32);
      s2 += __shfl_xor(s2, m, 32);
    }
    const float mean = s * (1.0f / 256.0f);
    const float var  = s2 * (1.0f / 256.0f) - mean * mean;
    const float inv  = rsqrtf(var + 1e-5f);

    float acc = 0.0f;
    #pragma unroll
    for (int r = 0; r < 8; ++r) {
      float g   = (h[r] - mean) * inv * gg[r] + bt[r];
      float cel = (g > 0.0f) ? g : (__expf(g) - 1.0f);   // CELU(alpha=1)
      acc += cel * w2[r];
    }
    #pragma unroll
    for (int m = 16; m >= 1; m >>= 1) acc += __shfl_xor(acc, m, 32);

    if (lane == 0) out[i * NN + j] = acc + b2s;
  }
}

extern "C" void kernel_launch(void* const* d_in, const int* in_sizes, int n_in,
                              void* d_out, int out_size, void* d_ws, size_t ws_size,
                              hipStream_t stream) {
  const float* xl    = (const float*)d_in[0];  // [512][64]
  const float* xr    = (const float*)d_in[1];  // [512][64]
  const float* W1    = (const float*)d_in[2];  // [384][256]
  const float* b1    = (const float*)d_in[3];  // [256]
  const float* gamma = (const float*)d_in[4];  // [256]
  const float* beta  = (const float*)d_in[5];  // [256]
  const float* W2    = (const float*)d_in[6];  // [256][1]
  const float* b2    = (const float*)d_in[7];  // [1]
  float* out = (float*)d_out;                  // [512][512]
  float* P   = (float*)d_ws;                   // 6 * 512 * 256 floats = 3.1 MB

  // Stage 1: 3072 WMMA tile-waves (6 GEMMs of 512x64 @ 64x256)
  fcpaired_stage1_wmma<<<768, 128, 0, stream>>>(xl, xr, W1, P);
  // Stage 2: 4096 waves over the 512x512 output grid
  fcpaired_stage2<<<512, 256, 0, stream>>>(P, b1, gamma, beta, W2, b2, out);
}
